// FRAB_EMAN_38001870635908
// MI455X (gfx1250) — compile-verified, matrix operand order checked by reading, not run on hardware
//
#include <hip/hip_runtime.h>
#include <hip/hip_bf16.h>

// ---------------------------------------------------------------------------
// Problem constants
// ---------------------------------------------------------------------------
#define BATCH 64
#define CIN   2048
#define HW    196
#define HWP   224          // HW padded (B-stage guard-free)
#define HID   4096
#define COUT  256
#define QDIM  32           // COUT / NUM_HEADS
#define QP    64           // QDIM padded to N-tile width
#define BN_EPS 1e-5f

typedef __attribute__((ext_vector_type(16))) __bf16 v16bf;
typedef __attribute__((ext_vector_type(8)))  __bf16 v8bf;
typedef __attribute__((ext_vector_type(8)))  float  v8f;

__device__ __forceinline__ float wave_sum(float v) {
    #pragma unroll
    for (int off = 16; off > 0; off >>= 1) v += __shfl_xor(v, off, 32);
    return v;
}
__device__ __forceinline__ float wave_max(float v) {
    #pragma unroll
    for (int off = 16; off > 0; off >>= 1) v = fmaxf(v, __shfl_xor(v, off, 32));
    return v;
}

// ---------------------------------------------------------------------------
// Bulk f32 -> bf16 weight conversion (n % 8 == 0), one b128 store / thread
// ---------------------------------------------------------------------------
__global__ __launch_bounds__(256) void cvt_kernel(const float* __restrict__ in,
                                                  __bf16* __restrict__ out, long n) {
    long i = ((long)blockIdx.x * 256 + threadIdx.x) * 8;
    if (i >= n) return;
    float4 f0 = *(const float4*)(in + i);
    float4 f1 = *(const float4*)(in + i + 4);
    v8bf o;
    o[0] = (__bf16)f0.x; o[1] = (__bf16)f0.y; o[2] = (__bf16)f0.z; o[3] = (__bf16)f0.w;
    o[4] = (__bf16)f1.x; o[5] = (__bf16)f1.y; o[6] = (__bf16)f1.z; o[7] = (__bf16)f1.w;
    *(v8bf*)(out + i) = o;
}

// x [rows][196] f32  ->  xbf [rows][224] bf16, zero padded
__global__ __launch_bounds__(256) void cvt_pad_kernel(const float* __restrict__ x,
                                                      __bf16* __restrict__ out) {
    long idx = (long)blockIdx.x * 256 + threadIdx.x;     // over rows*HWP
    int  l   = (int)(idx % HWP);
    long row = idx / HWP;
    float v  = (l < HW) ? x[row * HW + l] : 0.f;
    out[idx] = (__bf16)v;
}

// ---------------------------------------------------------------------------
// Pool over HW, output transposed bf16 [CIN][BATCH] (K-major B operand)
// ---------------------------------------------------------------------------
__global__ __launch_bounds__(256) void pool_kernel(const float* __restrict__ x,
                                                   __bf16* __restrict__ xpT) {
    int row  = blockIdx.x * 8 + (threadIdx.x >> 5);      // b*CIN + c
    int lane = threadIdx.x & 31;
    int b = row >> 11;
    int c = row & (CIN - 1);
    const float* p = x + (long)row * HW;
    float s = 0.f;
    for (int l = lane; l < HW; l += 32) s += p[l];
    s = wave_sum(s);
    if (lane == 0) xpT[c * BATCH + b] = (__bf16)(s * (1.0f / (float)HW));
}

// ---------------------------------------------------------------------------
// BN batch stats over (nb*L) samples of bf16 H[b][ch][Lrow]
// scale = g*rsqrt(var+eps), shift = beta - mean*scale
// ---------------------------------------------------------------------------
__global__ __launch_bounds__(256) void bn_stats_kernel(const __bf16* __restrict__ H,
                                                       const float* __restrict__ gamma,
                                                       const float* __restrict__ beta,
                                                       float* __restrict__ outScale,
                                                       float* __restrict__ outShift,
                                                       int nb, int L, int Lrow, long bStride) {
    __shared__ float rs[256], rs2[256];
    int ch = blockIdx.x;
    long total = (long)nb * L;
    float s = 0.f, s2 = 0.f;
    for (long i = threadIdx.x; i < total; i += 256) {
        int b = (int)(i / L);
        int l = (int)(i % L);
        float v = (float)H[(long)b * bStride + (long)ch * Lrow + l];
        s += v; s2 += v * v;
    }
    rs[threadIdx.x] = s; rs2[threadIdx.x] = s2;
    __syncthreads();
    #pragma unroll
    for (int off = 128; off > 0; off >>= 1) {
        if (threadIdx.x < off) {
            rs[threadIdx.x]  += rs[threadIdx.x + off];
            rs2[threadIdx.x] += rs2[threadIdx.x + off];
        }
        __syncthreads();
    }
    if (threadIdx.x == 0) {
        float inv_n = 1.0f / (float)total;
        float mean = rs[0] * inv_n;
        float var  = rs2[0] * inv_n - mean * mean;
        float sc   = gamma[ch] * rsqrtf(var + BN_EPS);
        outScale[ch] = sc;
        outShift[ch] = beta[ch] - mean * sc;
    }
}

// ---------------------------------------------------------------------------
// WMMA GEMM:  Out[b] = A[MxK] * B[b][KxLrow]   (all operands bf16)
//  * A fragments loaded DIRECTLY from global (row-major [M][K] matches the
//    16-bit A VGPR striping: lane = row, two contiguous 16B K-runs).
//  * B staged/transposed into double-buffered LDS (Bs[n][k]); BN_RELU fuses
//    relu(v*scale[k]+shift[k]) into staging (GEMM2 of each MLP).
//  * Block 256 thr = 8 waves; tile 128M x 64N, K-step 32.  Each wave owns a
//    16M x 64N strip: 1 A frag + 4 B frags -> 4 WMMA / wave / barrier.
//    All B fragments are hoisted into distinct locals BEFORE the WMMA chain
//    so the backend can clause the 8 ds_load_b128 and overlap LDS latency
//    under back-to-back WMMAs (instead of reusing one register quad).
//  * No guards on B staging (buffers padded; stray columns only feed output
//    columns the epilogue guard discards).
// ---------------------------------------------------------------------------
template <bool BN_RELU, bool OUT_BF16>
__global__ __launch_bounds__(256) void wmma_gemm_kernel(
        const __bf16* __restrict__ A,
        const __bf16* __restrict__ Bsrc,
        const float* __restrict__ bnScale,
        const float* __restrict__ bnShift,
        void* __restrict__ OutV,
        int M, int K, int Lrow, int Lw,            // Lw = output column write guard
        long strideB_b, long strideOut_b,
        int outRowStride, int outColStride) {
    __shared__ __bf16 Bs[2][64][40];               // +8 pad: 80B row stride

    const int tid  = threadIdx.x;
    const int lane = tid & 31;
    const int wave = tid >> 5;                     // 0..7 -> 16-row M strip
    const int l0 = blockIdx.x * 64;
    const int m0 = blockIdx.y * 128;
    const int bi = blockIdx.z;

    const __bf16* Bb = Bsrc + (long)bi * strideB_b;

    // B staging map: thread -> (k = tid&31, n-chunk of 8)
    const int sk = tid & 31;
    const int sn = (tid >> 5) * 8;

    // fragment addressing
    const int kb   = (lane < 16) ? 0 : 8;
    const int arow = m0 + wave * 16 + (lane & 15);
    const __bf16* Ap = A + (long)arow * K;
    const int nfr  = lane & 15;

    v8f acc[4] = {{}, {}, {}, {}};

    auto stage = [&](int k0, int buf) {
        float sc = 1.f, sh = 0.f;
        if constexpr (BN_RELU) { sc = bnScale[k0 + sk]; sh = bnShift[k0 + sk]; }
        v8bf raw = *(const v8bf*)(Bb + (long)(k0 + sk) * Lrow + l0 + sn);
        if (k0 + 64 < K)   // prefetch next-next B tile (global_prefetch_b8)
            __builtin_prefetch(Bb + (long)(k0 + 64 + sk) * Lrow + l0 + sn, 0, 1);
        #pragma unroll
        for (int i = 0; i < 8; ++i) {
            if constexpr (BN_RELU) {
                float v = fmaxf((float)raw[i] * sc + sh, 0.f);
                Bs[buf][sn + i][sk] = (__bf16)v;
            } else {
                Bs[buf][sn + i][sk] = raw[i];
            }
        }
    };

    stage(0, 0);
    __syncthreads();
    int buf = 0;
    for (int k0 = 0; k0 < K; k0 += 32) {
        if (k0 + 32 < K) stage(k0 + 32, buf ^ 1);

        v8bf a0 = *(const v8bf*)(Ap + k0 + kb);
        v8bf a1 = *(const v8bf*)(Ap + k0 + kb + 16);
        v16bf af = __builtin_shufflevector(a0, a1,
                     0,1,2,3,4,5,6,7,8,9,10,11,12,13,14,15);

        // hoist ALL B fragments into distinct registers first (clauses the
        // ds_loads; avoids per-WMMA s_wait_dscnt serialization)
        v16bf bfr[4];
        #pragma unroll
        for (int j = 0; j < 4; ++j) {
            v8bf b0 = *(const v8bf*)&Bs[buf][j * 16 + nfr][kb];
            v8bf b1 = *(const v8bf*)&Bs[buf][j * 16 + nfr][kb + 16];
            bfr[j] = __builtin_shufflevector(b0, b1,
                       0,1,2,3,4,5,6,7,8,9,10,11,12,13,14,15);
        }
        #pragma unroll
        for (int j = 0; j < 4; ++j)
            acc[j] = __builtin_amdgcn_wmma_f32_16x16x32_bf16(
                         false, af, false, bfr[j], (short)0, acc[j], false, false);

        __syncthreads();
        buf ^= 1;
    }

    // epilogue: D layout -> lane = N + 16*(M>=8), VGPR r = M&7
    const int mbase = m0 + wave * 16 + ((lane < 16) ? 0 : 8);
    #pragma unroll
    for (int j = 0; j < 4; ++j) {
        int n = l0 + j * 16 + nfr;
        if (n < Lw) {
            if constexpr (OUT_BF16) {
                __bf16* op = (__bf16*)OutV + (long)bi * strideOut_b;
                #pragma unroll
                for (int r = 0; r < 8; ++r)
                    op[(long)(mbase + r) * outRowStride + n] = (__bf16)acc[j][r];
            } else {
                float* op = (float*)OutV + (long)bi * strideOut_b;
                #pragma unroll
                for (int r = 0; r < 8; ++r)
                    op[(long)(mbase + r) * outRowStride + (long)n * outColStride] = acc[j][r];
            }
        }
    }
}

// ---------------------------------------------------------------------------
// Channel-L2 norm of z_feat (f32, row stride HWP) at each (b,l)
// ---------------------------------------------------------------------------
__global__ __launch_bounds__(256) void norm_kernel(const float* __restrict__ zf,
                                                   float* __restrict__ nrm) {
    int b = blockIdx.x;
    int l = threadIdx.x;
    if (l >= HW) return;
    const float* p = zf + (long)b * COUT * HWP + l;
    float s = 0.f;
    #pragma unroll 4
    for (int c = 0; c < COUT; ++c) { float v = p[c * HWP]; s += v * v; }
    nrm[b * HWP + l] = fmaxf(sqrtf(s), 1e-12f);      // SCALE == 1.0
}

// ---------------------------------------------------------------------------
// In-place softmax over l (196) per (b, channel) row; one wave per row
// ---------------------------------------------------------------------------
__global__ __launch_bounds__(256) void softmax_kernel(float* __restrict__ zf,
                                                      const float* __restrict__ nrm) {
    int row  = blockIdx.x * 8 + (threadIdx.x >> 5);  // b*COUT + c
    int lane = threadIdx.x & 31;
    int b = row >> 8;
    float* p = zf + (long)row * HWP;
    const float* nr = nrm + b * HWP;

    float vals[7];
    float mx = -__builtin_inff();
    #pragma unroll
    for (int i = 0; i < 7; ++i) {
        int l = lane + i * 32;
        float v = (l < HW) ? p[l] / nr[l] : -__builtin_inff();
        vals[i] = v;
        mx = fmaxf(mx, v);
    }
    mx = wave_max(mx);
    float sum = 0.f;
    #pragma unroll
    for (int i = 0; i < 7; ++i) {
        int l = lane + i * 32;
        float e = (l < HW) ? __expf(vals[i] - mx) : 0.f;
        vals[i] = e; sum += e;
    }
    sum = wave_sum(sum);
    float inv = 1.0f / sum;
    #pragma unroll
    for (int i = 0; i < 7; ++i) {
        int l = lane + i * 32;
        if (l < HW) p[l] = vals[i] * inv;
    }
}

// ---------------------------------------------------------------------------
// obj_pre[b][c][q] = sum_l x[b][c][l] * attn[b][(c/256)*32+q][l]
// writes bf16 padded to QP=64 columns (q>=32 -> 0) so the next GEMM is
// guard-free.
// ---------------------------------------------------------------------------
__global__ __launch_bounds__(256) void attnval_kernel(const float* __restrict__ x,
                                                      const float* __restrict__ attn,
                                                      __bf16* __restrict__ objpre) {
    long idx = (long)blockIdx.x * 256 + threadIdx.x;  // over b*CIN*QP
    int q = (int)(idx & (QP - 1));
    int c = (int)((idx >> 6) & (CIN - 1));
    int b = (int)(idx >> 17);                         // CIN*QP = 131072
    float s = 0.f;
    if (q < QDIM) {
        const float* xp = x    + ((long)b * CIN + c) * HW;
        const float* ap = attn + ((long)b * COUT + (c >> 8) * QDIM + q) * HWP;
        #pragma unroll 4
        for (int l = 0; l < HW; ++l) s += xp[l] * ap[l];
    }
    objpre[idx] = (__bf16)s;
}

// ---------------------------------------------------------------------------
// Host launcher
// ---------------------------------------------------------------------------
extern "C" void kernel_launch(void* const* d_in, const int* in_sizes, int n_in,
                              void* d_out, int out_size, void* d_ws, size_t ws_size,
                              hipStream_t stream) {
    const float* x    = (const float*)d_in[0];
    const float* pW1  = (const float*)d_in[1];
    const float* pG1  = (const float*)d_in[2];
    const float* pB1  = (const float*)d_in[3];
    const float* pW2  = (const float*)d_in[4];
    const float* ppW1 = (const float*)d_in[5];
    const float* ppG1 = (const float*)d_in[6];
    const float* ppB1 = (const float*)d_in[7];
    const float* ppW2 = (const float*)d_in[8];
    const float* poW1 = (const float*)d_in[9];
    const float* poG1 = (const float*)d_in[10];
    const float* poB1 = (const float*)d_in[11];
    const float* poW2 = (const float*)d_in[12];
    (void)in_sizes; (void)n_in; (void)out_size; (void)ws_size;

    float* out_zg  = (float*)d_out;                   // [64][256]
    float* out_obj = (float*)d_out + BATCH * COUT;    // [64][256][32]

    // ---- workspace carve-up (16B aligned chunks) ----------------------------
    char* w = (char*)d_ws;
    auto take = [&](long bytes) { char* p = w; w += (bytes + 15) & ~15L; return p; };
    __bf16* xbf    = (__bf16*)take((long)BATCH * CIN * HWP * 2);   // padded x
    __bf16* pW1b   = (__bf16*)take((long)HID * CIN * 2);
    __bf16* pW2b   = (__bf16*)take((long)COUT * HID * 2);
    __bf16* ppW1b  = (__bf16*)take((long)HID * CIN * 2);
    __bf16* ppW2b  = (__bf16*)take((long)COUT * HID * 2);
    __bf16* poW1b  = (__bf16*)take((long)HID * CIN * 2);
    __bf16* poW2b  = (__bf16*)take((long)COUT * HID * 2);
    __bf16* xpT    = (__bf16*)take((long)CIN * BATCH * 2);         // [2048][64]
    __bf16* h_g    = (__bf16*)take((long)HID * BATCH * 2);         // [4096][64]
    __bf16* h_f    = (__bf16*)take((long)BATCH * HID * HWP * 2);   // [64][4096][224]
    __bf16* objpre = (__bf16*)take((long)BATCH * CIN * QP * 2);    // [64][2048][64]
    __bf16* h_o    = (__bf16*)take((long)BATCH * HID * QP * 2);    // [64][4096][64]
    float*  z_feat = (float*)take((long)BATCH * COUT * HWP * 4);   // [64][256][224]
    float*  nrm    = (float*)take((long)BATCH * HWP * 4);
    float*  sc_g   = (float*)take(HID * 4);
    float*  sh_g   = (float*)take(HID * 4);
    float*  sc_f   = (float*)take(HID * 4);
    float*  sh_f   = (float*)take(HID * 4);
    float*  sc_o   = (float*)take(HID * 4);
    float*  sh_o   = (float*)take(HID * 4);
    (void)take(256);                                               // over-read slack

    dim3 blk(256);

    // ---- one-time conversions ----------------------------------------------
    cvt_pad_kernel<<<(BATCH * CIN * HWP) / 256, blk, 0, stream>>>(x, xbf);
    long wBig = (long)HID * CIN, wSml = (long)COUT * HID;
    cvt_kernel<<<(int)(wBig / 8 / 256), blk, 0, stream>>>(pW1,  pW1b,  wBig);
    cvt_kernel<<<(int)(wSml / 8 / 256), blk, 0, stream>>>(pW2,  pW2b,  wSml);
    cvt_kernel<<<(int)(wBig / 8 / 256), blk, 0, stream>>>(ppW1, ppW1b, wBig);
    cvt_kernel<<<(int)(wSml / 8 / 256), blk, 0, stream>>>(ppW2, ppW2b, wSml);
    cvt_kernel<<<(int)(wBig / 8 / 256), blk, 0, stream>>>(poW1, poW1b, wBig);
    cvt_kernel<<<(int)(wSml / 8 / 256), blk, 0, stream>>>(poW2, poW2b, wSml);
    pool_kernel<<<(BATCH * CIN) / 8, blk, 0, stream>>>(x, xpT);

    // ---- pooled (z_g) branch ------------------------------------------------
    wmma_gemm_kernel<false, true><<<dim3(1, HID / 128, 1), blk, 0, stream>>>(
        pW1b, xpT, nullptr, nullptr, h_g,
        HID, CIN, BATCH, BATCH, 0L, 0L, BATCH, 1);
    bn_stats_kernel<<<HID, blk, 0, stream>>>(h_g, pG1, pB1, sc_g, sh_g,
                                             1, BATCH, BATCH, 0L);
    wmma_gemm_kernel<true, false><<<dim3(1, COUT / 128, 1), blk, 0, stream>>>(
        pW2b, h_g, sc_g, sh_g, out_zg,
        COUT, HID, BATCH, BATCH, 0L, 0L, /*row*/1, /*col*/COUT);   // transposed

    // ---- z_feat branch (dominant) ------------------------------------------
    wmma_gemm_kernel<false, true><<<dim3(HWP / 64 + 1, HID / 128, BATCH), blk, 0, stream>>>(
        ppW1b, xbf, nullptr, nullptr, h_f,
        HID, CIN, HWP, HWP, (long)CIN * HWP, (long)HID * HWP, HWP, 1);
    bn_stats_kernel<<<HID, blk, 0, stream>>>(h_f, ppG1, ppB1, sc_f, sh_f,
                                             BATCH, HW, HWP, (long)HID * HWP);
    wmma_gemm_kernel<true, false><<<dim3(HWP / 64 + 1, COUT / 128, BATCH), blk, 0, stream>>>(
        ppW2b, h_f, sc_f, sh_f, z_feat,
        COUT, HID, HWP, HW, (long)HID * HWP, (long)COUT * HWP, HWP, 1);

    // ---- attention ----------------------------------------------------------
    norm_kernel<<<BATCH, blk, 0, stream>>>(z_feat, nrm);
    softmax_kernel<<<(BATCH * COUT) / 8, blk, 0, stream>>>(z_feat, nrm);
    attnval_kernel<<<(BATCH * CIN * QP) / 256, blk, 0, stream>>>(x, z_feat, objpre);

    // ---- obj MLP ------------------------------------------------------------
    wmma_gemm_kernel<false, true><<<dim3(1, HID / 128, BATCH), blk, 0, stream>>>(
        poW1b, objpre, nullptr, nullptr, h_o,
        HID, CIN, QP, QP, (long)CIN * QP, (long)HID * QP, QP, 1);
    bn_stats_kernel<<<HID, blk, 0, stream>>>(h_o, poG1, poB1, sc_o, sh_o,
                                             BATCH, QDIM, QP, (long)HID * QP);
    wmma_gemm_kernel<true, false><<<dim3(1, COUT / 128, BATCH), blk, 0, stream>>>(
        poW2b, h_o, sc_o, sh_o, out_obj,
        COUT, HID, QP, QDIM, (long)HID * QP, (long)COUT * QDIM, QDIM, 1);
}
